// MambaSequenceModel_17394617548839
// MI455X (gfx1250) — compile-verified, matrix operand order checked by reading, not run on hardware
//
#include <hip/hip_runtime.h>

// ---------------- problem constants ----------------
constexpr int Bsz = 2, Lseq = 1024, Hdim = 512;
constexpr int DI  = 2 * Hdim;        // 1024
constexpr int NS  = 8;               // SSM state dim
constexpr int KC  = 4;               // depthwise conv taps
constexpr int Rrk = 32;              // dt rank
constexpr int NLayers = 4;
constexpr int VMAXc = 300;
constexpr float EPSc = 1e-5f;
constexpr int SPW = 80;              // padded row width for sp (R+2N=48 -> 80)
constexpr int CLc = 64, NCh = 16;    // scan chunking: 16 chunks of 64 steps
constexpr int BLH = Bsz * Lseq;      // 2048 rows

typedef __attribute__((ext_vector_type(16))) __bf16 v16bf;
typedef __attribute__((ext_vector_type(8)))  __bf16 v8bf;
typedef __attribute__((ext_vector_type(8)))  float  v8f;

__device__ __forceinline__ float sigmoidf_(float x) { return 1.f / (1.f + __expf(-x)); }

// Load a 16x32 bf16 fragment (A or B operand) in the CDNA5 wave32 layout:
// lane<16 holds K {0..7, 16..23} of row (lane&15); lane>=16 holds K {8..15, 24..31}.
// Caller pre-offsets p by (row*stride + (lane>>4)*8).
__device__ __forceinline__ v16bf ld_frag(const __bf16* p) {
  union { v16bf v; v8bf h[2]; } u;
  u.h[0] = *(const v8bf*)(p);
  u.h[1] = *(const v8bf*)(p + 16);
  return u.v;
}

// CDNA5 async LDS-DMA: 16B per lane straight from global into LDS (no VGPR
// round trip), tracked by ASYNCcnt.
__device__ __forceinline__ void async_copy_b128(unsigned lds_addr, const void* gaddr) {
  asm volatile("global_load_async_to_lds_b128 %0, %1, off"
               :: "v"(lds_addr), "v"(gaddr) : "memory");
}
__device__ __forceinline__ void wait_async0() {
  asm volatile("s_wait_asynccnt 0x0" ::: "memory");
}

// ---------------- bf16 WMMA GEMM: C[M,N] = A[M,K] * W[N,K]^T (+epilogue) ----------------
// B tile is double-buffered in LDS and filled with global_load_async_to_lds_b128,
// overlapping the k+1 copy with the 4 WMMAs of step k.
// mode 0: store   1: store + bias[n]   2: softplus(x + bias[n])   3: C[m,n] += x
__global__ __launch_bounds__(256)
void gemm_bf16_wmma(const __bf16* __restrict__ A, int lda,
                    const __bf16* __restrict__ W, int ldw,
                    float* __restrict__ C, int ldc,
                    const float* __restrict__ bias,
                    int M, int N, int K, int mode) {
  __shared__ __bf16 Bs[2][64 * 32];           // 2 x (64 weight rows x 32 k)
  const int tid  = threadIdx.x;
  const int lane = tid & 31;
  const int wave = tid >> 5;
  const int r    = lane & 15;
  const int half = lane >> 4;
  const int m0   = blockIdx.x * 128 + wave * 16;
  const int n0   = blockIdx.y * 64;

  const __bf16* arow = A + (size_t)(m0 + r) * lda + half * 8;

  // per-thread 16B slice of the 64x32 weight tile
  const int nl = tid >> 2;
  const int kk = (tid & 3) * 8;
  int nb = n0 + nl; if (nb >= N) nb = N - 1;
  const __bf16* wrow = W + (size_t)nb * ldw + kk;
  const unsigned lds0 = (unsigned)(size_t)(void*)&Bs[0][nl * 32 + kk];
  const unsigned lds1 = (unsigned)(size_t)(void*)&Bs[1][nl * 32 + kk];

  const int T = K / 32;
  async_copy_b128(lds0, wrow);                // prologue: tile 0 -> buf 0
  v8f acc[4] = {};
  for (int kt = 0; kt < T; ++kt) {
    wait_async0();                            // own slice of tile kt arrived
    __syncthreads();                          // whole tile kt visible; buf[kt+1] free
    if (kt + 1 < T)
      async_copy_b128((kt & 1) ? lds0 : lds1, wrow + (kt + 1) * 32);
    if (kt + 2 < T) __builtin_prefetch(arow + (size_t)(kt + 2) * 32, 0, 0);
    v16bf afrag = ld_frag(arow + (size_t)kt * 32);
    const __bf16* bbase = &Bs[kt & 1][0];
#pragma unroll
    for (int j = 0; j < 4; ++j) {
      v16bf bfrag = ld_frag(&bbase[(16 * j + r) * 32 + half * 8]);
      acc[j] = __builtin_amdgcn_wmma_f32_16x16x32_bf16(
          false, afrag, false, bfrag, (short)0, acc[j], false, false);
    }
  }
  // epilogue: acc[j][g] -> (m0 + g + half*8, n0 + 16j + r)
#pragma unroll
  for (int j = 0; j < 4; ++j) {
    int n = n0 + 16 * j + r;
    if (n >= N) continue;
    float bv = (mode == 1 || mode == 2) ? bias[n] : 0.f;
#pragma unroll
    for (int g = 0; g < 8; ++g) {
      int m = m0 + g + half * 8;
      float v = acc[j][g];
      if (mode == 1) v += bv;
      else if (mode == 2) { v += bv; v = (v > 20.f) ? v : log1pf(__expf(v)); }
      float* cp = &C[(size_t)m * ldc + n];
      if (mode == 3) *cp += v; else *cp = v;
    }
  }
}

// ---------------- elementwise / setup kernels ----------------
__global__ void f32_to_bf16_kern(const float* __restrict__ in, __bf16* __restrict__ out, int n) {
  int i = blockIdx.x * 256 + threadIdx.x;
  if (i < n) out[i] = (__bf16)in[i];
}

__global__ void embed_kern(const int* __restrict__ x, const float* __restrict__ emb,
                           float* __restrict__ h) {
  int i = blockIdx.x * 256 + threadIdx.x;     // over B*L*H
  int hh = i % Hdim, bl = i / Hdim;
  float v = 0.f;
#pragma unroll
  for (int f = 0; f < 4; ++f) {
    int tok = x[bl * 4 + f];
    v += emb[((size_t)f * VMAXc + tok) * Hdim + hh];
  }
  h[i] = v;
}

__global__ __launch_bounds__(256)
void rmsnorm_bf16_kern(const float* __restrict__ h, const float* __restrict__ w,
                       __bf16* __restrict__ out) {
  __shared__ float red[8];
  int bl = blockIdx.x, tid = threadIdx.x;
  const float* row = h + (size_t)bl * Hdim;
  float ss = 0.f;
  for (int i = tid; i < Hdim; i += 256) { float v = row[i]; ss += v * v; }
  for (int off = 16; off; off >>= 1) ss += __shfl_down(ss, off, 32);
  if ((tid & 31) == 0) red[tid >> 5] = ss;
  __syncthreads();
  if (tid == 0) {
    float s = 0.f;
#pragma unroll
    for (int i = 0; i < 8; ++i) s += red[i];
    red[0] = rsqrtf(s / Hdim + EPSc);
  }
  __syncthreads();
  float rs = red[0];
  for (int i = tid; i < Hdim; i += 256) out[(size_t)bl * Hdim + i] = (__bf16)(row[i] * rs * w[i]);
}

// causal depthwise conv (K=4) + SiLU over proj[:, :DI]; writes f32 + bf16
__global__ void conv_silu_kern(const float* __restrict__ proj, const float* __restrict__ cw,
                               const float* __restrict__ cb, float* __restrict__ hsf,
                               __bf16* __restrict__ hsb) {
  int i = blockIdx.x * 256 + threadIdx.x;     // over B*L*DI
  int d = i % DI, bl = i / DI, l = bl % Lseq;
  const float* w = cw + (size_t)d * KC;
  float acc = cb[d];
#pragma unroll
  for (int j = 0; j < KC; ++j) {
    int ll = l - 3 + j;
    if (ll >= 0) acc += w[j] * proj[(size_t)(bl - (3 - j)) * (2 * DI) + d];
  }
  acc = acc * sigmoidf_(acc);
  hsf[i] = acc;
  hsb[i] = (__bf16)acc;
}

__global__ void extract_dtr_kern(const float* __restrict__ sp, __bf16* __restrict__ dtr) {
  int i = blockIdx.x * 256 + threadIdx.x;     // over B*L*R
  int r = i & 31, row = i >> 5;
  dtr[i] = (__bf16)sp[(size_t)row * SPW + r];
}

// ---------------- chunked linear-recurrence scan ----------------
// pass 1: per-chunk (P = prod dA, S = local final state) with zero init
__global__ __launch_bounds__(256)
void scan_chunk_reduce(const float* __restrict__ dt, const float* __restrict__ hs,
                       const float* __restrict__ sp, const float* __restrict__ A_log,
                       float* __restrict__ Pout, float* __restrict__ Sout) {
  __shared__ float sB[CLc * NS];
  int d = blockIdx.x * 256 + threadIdx.x;
  int c = blockIdx.y, b = blockIdx.z;
  int rowbase = b * Lseq + c * CLc;
  for (int i = threadIdx.x; i < CLc * NS; i += 256)
    sB[i] = sp[(size_t)(rowbase + (i >> 3)) * SPW + Rrk + (i & 7)];
  __syncthreads();
  float An[NS], Pv[NS], Sv[NS];
#pragma unroll
  for (int n = 0; n < NS; ++n) {
    An[n] = -__expf(A_log[(size_t)d * NS + n]);
    Pv[n] = 1.f; Sv[n] = 0.f;
  }
  for (int t = 0; t < CLc; ++t) {
    size_t ix = (size_t)(rowbase + t) * DI + d;
    float dtv = dt[ix], u = dtv * hs[ix];
#pragma unroll
    for (int n = 0; n < NS; ++n) {
      float a = __expf(dtv * An[n]);
      Sv[n] = a * Sv[n] + u * sB[t * NS + n];
      Pv[n] *= a;
    }
  }
  size_t base = ((size_t)(b * DI + d) * NS) * NCh + c;
#pragma unroll
  for (int n = 0; n < NS; ++n) { Pout[base + (size_t)n * NCh] = Pv[n]; Sout[base + (size_t)n * NCh] = Sv[n]; }
}

// pass 2: serial carry across 16 chunks per (b,d,n)
__global__ void scan_carry(const float* __restrict__ P, const float* __restrict__ S,
                           float* __restrict__ Cinit) {
  int i = blockIdx.x * 256 + threadIdx.x;     // over B*DI*NS
  size_t base = (size_t)i * NCh;
  float carry = 0.f;
  for (int c = 0; c < NCh; ++c) {
    Cinit[base + c] = carry;
    carry = P[base + c] * carry + S[base + c];
  }
}

// pass 3: replay with correct init; fuse y += hs*D and silu(gate); emit bf16
__global__ __launch_bounds__(256)
void scan_final(const float* __restrict__ dt, const float* __restrict__ hs,
                const float* __restrict__ sp, const float* __restrict__ A_log,
                const float* __restrict__ Dssm, const float* __restrict__ proj,
                const float* __restrict__ Cinit, __bf16* __restrict__ ybf) {
  __shared__ float sB[CLc * NS], sC[CLc * NS];
  int d = blockIdx.x * 256 + threadIdx.x;
  int c = blockIdx.y, b = blockIdx.z;
  int rowbase = b * Lseq + c * CLc;
  for (int i = threadIdx.x; i < CLc * NS; i += 256) {
    const float* sprow = sp + (size_t)(rowbase + (i >> 3)) * SPW + Rrk;
    sB[i] = sprow[i & 7];
    sC[i] = sprow[NS + (i & 7)];
  }
  __syncthreads();
  float An[NS], st[NS];
  size_t base = ((size_t)(b * DI + d) * NS) * NCh + c;
#pragma unroll
  for (int n = 0; n < NS; ++n) {
    An[n] = -__expf(A_log[(size_t)d * NS + n]);
    st[n] = Cinit[base + (size_t)n * NCh];
  }
  float Dv = Dssm[d];
  for (int t = 0; t < CLc; ++t) {
    size_t ix = (size_t)(rowbase + t) * DI + d;
    float dtv = dt[ix], hsv = hs[ix], u = dtv * hsv;
    float y = 0.f;
#pragma unroll
    for (int n = 0; n < NS; ++n) {
      float a = __expf(dtv * An[n]);
      st[n] = a * st[n] + u * sB[t * NS + n];
      y += st[n] * sC[t * NS + n];
    }
    y += hsv * Dv;
    float g = proj[(size_t)(rowbase + t) * (2 * DI) + DI + d];
    y *= g * sigmoidf_(g);
    ybf[ix] = (__bf16)y;
  }
}

// ---------------- host orchestration ----------------
extern "C" void kernel_launch(void* const* d_in, const int* in_sizes, int n_in,
                              void* d_out, int out_size, void* d_ws, size_t ws_size,
                              hipStream_t stream) {
  (void)in_sizes; (void)n_in; (void)out_size; (void)ws_size;
  const int*   x        = (const int*)  d_in[0];
  const float* emb      = (const float*)d_in[1];
  const float* in_w     = (const float*)d_in[2];
  const float* conv_w   = (const float*)d_in[3];
  const float* conv_b   = (const float*)d_in[4];
  const float* x_w      = (const float*)d_in[5];
  const float* dt_w     = (const float*)d_in[6];
  const float* dt_b     = (const float*)d_in[7];
  const float* A_log    = (const float*)d_in[8];
  const float* D_ssm    = (const float*)d_in[9];
  const float* out_w    = (const float*)d_in[10];
  const float* norm_w   = (const float*)d_in[11];
  const float* norm_f_w = (const float*)d_in[12];
  const float* head_w   = (const float*)d_in[13];
  const float* head_b   = (const float*)d_in[14];
  float* outp = (float*)d_out;

  char* wp = (char*)d_ws;
  auto alloc = [&](size_t bytes) -> char* {
    char* r = wp; wp += (bytes + 255) & ~(size_t)255; return r;
  };
  const size_t n_in_w  = (size_t)NLayers * 2 * DI * Hdim;
  const size_t n_x_w   = (size_t)NLayers * (Rrk + 2 * NS) * DI;
  const size_t n_dt_w  = (size_t)NLayers * DI * Rrk;
  const size_t n_out_w = (size_t)NLayers * Hdim * DI;
  const size_t n_head  = (size_t)VMAXc * Hdim;

  __bf16* wb_in   = (__bf16*)alloc(n_in_w  * 2);
  __bf16* wb_x    = (__bf16*)alloc(n_x_w   * 2);
  __bf16* wb_dt   = (__bf16*)alloc(n_dt_w  * 2);
  __bf16* wb_out  = (__bf16*)alloc(n_out_w * 2);
  __bf16* wb_head = (__bf16*)alloc(n_head  * 2);
  float*  hbuf    = (float*) alloc((size_t)BLH * Hdim * 4);
  __bf16* hn      = (__bf16*)alloc((size_t)BLH * Hdim * 2);
  float*  proj    = (float*) alloc((size_t)BLH * 2 * DI * 4);
  float*  hsf     = (float*) alloc((size_t)BLH * DI * 4);
  __bf16* hsb     = (__bf16*)alloc((size_t)BLH * DI * 2);
  float*  spf     = (float*) alloc((size_t)BLH * SPW * 4);
  __bf16* dtr     = (__bf16*)alloc((size_t)BLH * Rrk * 2);
  float*  dtf     = (float*) alloc((size_t)BLH * DI * 4);
  __bf16* ybf     = (__bf16*)alloc((size_t)BLH * DI * 2);
  float*  Pbuf    = (float*) alloc((size_t)Bsz * DI * NS * NCh * 4);
  float*  Sbuf    = (float*) alloc((size_t)Bsz * DI * NS * NCh * 4);
  float*  Cini    = (float*) alloc((size_t)Bsz * DI * NS * NCh * 4);

  const dim3 blk(256);
  auto cdiv = [](size_t a, size_t b) { return (unsigned)((a + b - 1) / b); };

  // one-time (per launch) fp32 -> bf16 weight conversion
  f32_to_bf16_kern<<<cdiv(n_in_w, 256),  blk, 0, stream>>>(in_w,   wb_in,   (int)n_in_w);
  f32_to_bf16_kern<<<cdiv(n_x_w, 256),   blk, 0, stream>>>(x_w,    wb_x,    (int)n_x_w);
  f32_to_bf16_kern<<<cdiv(n_dt_w, 256),  blk, 0, stream>>>(dt_w,   wb_dt,   (int)n_dt_w);
  f32_to_bf16_kern<<<cdiv(n_out_w, 256), blk, 0, stream>>>(out_w,  wb_out,  (int)n_out_w);
  f32_to_bf16_kern<<<cdiv(n_head, 256),  blk, 0, stream>>>(head_w, wb_head, (int)n_head);

  // token embedding gather-sum
  embed_kern<<<cdiv((size_t)BLH * Hdim, 256), blk, 0, stream>>>(x, emb, hbuf);

  const dim3 g_scan(DI / 256, NCh, Bsz);
  for (int l = 0; l < NLayers; ++l) {
    // RMSNorm -> bf16
    rmsnorm_bf16_kern<<<BLH, blk, 0, stream>>>(hbuf, norm_w + (size_t)l * Hdim, hn);
    // in_proj: (2048,512) x (2048,512)^T -> proj (2048,2048)
    gemm_bf16_wmma<<<dim3(BLH / 128, 2 * DI / 64), blk, 0, stream>>>(
        hn, Hdim, wb_in + (size_t)l * 2 * DI * Hdim, Hdim, proj, 2 * DI,
        nullptr, BLH, 2 * DI, Hdim, 0);
    // depthwise causal conv + SiLU
    conv_silu_kern<<<cdiv((size_t)BLH * DI, 256), blk, 0, stream>>>(
        proj, conv_w + (size_t)l * DI * KC, conv_b + (size_t)l * DI, hsf, hsb);
    // x_proj: (2048,1024) x (48,1024)^T -> sp (2048,80-padded)
    gemm_bf16_wmma<<<dim3(BLH / 128, 2), blk, 0, stream>>>(
        hsb, DI, wb_x + (size_t)l * (Rrk + 2 * NS) * DI, DI, spf, SPW,
        nullptr, BLH, Rrk + 2 * NS, DI, 0);
    // dt low-rank expand + softplus
    extract_dtr_kern<<<cdiv((size_t)BLH * Rrk, 256), blk, 0, stream>>>(spf, dtr);
    gemm_bf16_wmma<<<dim3(BLH / 128, DI / 64), blk, 0, stream>>>(
        dtr, Rrk, wb_dt + (size_t)l * DI * Rrk, Rrk, dtf, DI,
        dt_b + (size_t)l * DI, BLH, DI, Rrk, 2);
    // chunked selective-scan
    scan_chunk_reduce<<<g_scan, blk, 0, stream>>>(
        dtf, hsf, spf, A_log + (size_t)l * DI * NS, Pbuf, Sbuf);
    scan_carry<<<Bsz * DI * NS / 256, blk, 0, stream>>>(Pbuf, Sbuf, Cini);
    scan_final<<<g_scan, blk, 0, stream>>>(
        dtf, hsf, spf, A_log + (size_t)l * DI * NS, D_ssm + (size_t)l * DI,
        proj, Cini, ybf);
    // out_proj, residual-accumulated into h
    gemm_bf16_wmma<<<dim3(BLH / 128, Hdim / 64), blk, 0, stream>>>(
        ybf, DI, wb_out + (size_t)l * Hdim * DI, DI, hbuf, Hdim,
        nullptr, BLH, Hdim, DI, 3);
  }
  // final norm + LM head (bias fused)
  rmsnorm_bf16_kern<<<BLH, blk, 0, stream>>>(hbuf, norm_f_w, hn);
  gemm_bf16_wmma<<<dim3(BLH / 128, (VMAXc + 63) / 64), blk, 0, stream>>>(
      hn, Hdim, wb_head, Hdim, outp, VMAXc, head_b, BLH, VMAXc, Hdim, 1);
}
